// Attention_65962107732223
// MI455X (gfx1250) — matrix operand order repaired and checked by reading, not compiled, not run on hardware
//
#include <hip/hip_runtime.h>

// ---------------------------------------------------------------------------
// Problem constants (from reference): B=2, T=2048, E=1024, H=16, D=64
// ---------------------------------------------------------------------------
#define BB 2
#define TT 2048
#define EE 1024
#define HH 16
#define DD 64
#define FF (3 * EE)      // 3072
#define MM (BB * TT)     // 4096 rows for both GEMMs

typedef __attribute__((ext_vector_type(16))) __bf16 v16bf;
typedef __attribute__((ext_vector_type(8)))  float  v8f;

union Frag { unsigned int u[8]; v16bf v; };

__device__ __forceinline__ unsigned short f2bf(float f) {
    unsigned int u = __float_as_uint(f);
    unsigned int r = u + 0x7FFFu + ((u >> 16) & 1u);   // round-to-nearest-even
    return (unsigned short)(r >> 16);
}

// ---------------------------------------------------------------------------
// fp32 -> bf16 elementwise convert
// ---------------------------------------------------------------------------
__global__ void cvt_bf16_kernel(const float* __restrict__ src,
                                unsigned short* __restrict__ dst, int n) {
    int i = blockIdx.x * blockDim.x + threadIdx.x;
    if (i < n) dst[i] = f2bf(src[i]);
}

// ---------------------------------------------------------------------------
// WMMA GEMM:  C[M,N] (f32) = A[M,K] (bf16, row-major) * Bw[N,K]^T (bf16)
// One wave computes a 64x64 tile (4x4 grid of 16x16 WMMA tiles):
// 16 WMMAs per 32-wide K-step against 16 b128 fragment loads (4x reuse).
// Fragment loads follow the CDNA5 16-bit A(16x32) / B(32x16) VGPR layouts.
// ---------------------------------------------------------------------------
__global__ __launch_bounds__(256) void gemm_bf16_kernel(
    const unsigned short* __restrict__ A,
    const unsigned short* __restrict__ Bw,
    float* __restrict__ C, int M, int N, int K) {
    const int lane  = threadIdx.x & 31;
    const int wave  = threadIdx.x >> 5;
    const int wid   = blockIdx.x * (blockDim.x >> 5) + wave;
    const int tilesN = N >> 6;
    const int m0 = (wid / tilesN) << 6;
    const int n0 = (wid % tilesN) << 6;
    const int khalf = lane >> 4;     // 0: lanes 0-15, 1: lanes 16-31
    const int lm    = lane & 15;

    v8f acc[4][4] = {};

    for (int k0 = 0; k0 < K; k0 += 32) {
        Frag a[4], b[4];
#pragma unroll
        for (int mi = 0; mi < 4; ++mi) {
            const unsigned short* base = A + (size_t)(m0 + mi * 16 + lm) * K;
#pragma unroll
            for (int j = 0; j < 8; ++j) {
                int k = k0 + ((j & 3) * 2) + ((j >> 2) * 16) + khalf * 8;
                a[mi].u[j] = *(const unsigned int*)(base + k);
            }
        }
#pragma unroll
        for (int ni = 0; ni < 4; ++ni) {
            const unsigned short* base = Bw + (size_t)(n0 + ni * 16 + lm) * K;
#pragma unroll
            for (int j = 0; j < 8; ++j) {
                int k = k0 + ((j & 3) * 2) + ((j >> 2) * 16) + khalf * 8;
                b[ni].u[j] = *(const unsigned int*)(base + k);
            }
        }
#pragma unroll
        for (int mi = 0; mi < 4; ++mi)
#pragma unroll
            for (int ni = 0; ni < 4; ++ni)
                acc[mi][ni] = __builtin_amdgcn_wmma_f32_16x16x32_bf16(
                    false, a[mi].v, false, b[ni].v, (short)0, acc[mi][ni],
                    false, false);
    }

#pragma unroll
    for (int mi = 0; mi < 4; ++mi)
#pragma unroll
        for (int ni = 0; ni < 4; ++ni)
#pragma unroll
            for (int r = 0; r < 8; ++r) {
                int row = m0 + mi * 16 + r + khalf * 8;
                int col = n0 + ni * 16 + lm;
                C[(size_t)row * N + col] = acc[mi][ni][r];
            }
}

// ---------------------------------------------------------------------------
// RoPE + head split.
// qkv f32 [B,T,3,H,D]  ->  qb,kb bf16 [B,H,T,D] (rope applied),
//                          vb bf16 [B,H,D,T] (transposed for P*V B-frags).
// Each thread handles one rotation pair (i, i+32), i in [0,32).
// ---------------------------------------------------------------------------
__global__ void rope_split_kernel(const float* __restrict__ qkv,
                                  unsigned short* __restrict__ qb,
                                  unsigned short* __restrict__ kb,
                                  unsigned short* __restrict__ vb) {
    int tid = blockIdx.x * blockDim.x + threadIdx.x;  // B*T*H*32 total
    int i = tid & 31;
    int r = tid >> 5;
    int h = r & (HH - 1); r >>= 4;
    int t = r & (TT - 1);
    int b = r >> 11;

    const float* base = qkv + (size_t)(b * TT + t) * FF + h * DD;

    float c, s;
    if (i < 16) {
        float fr = __powf(10000.0f, -(float)i / 16.0f);
        float th = (float)t * fr;
        c = __cosf(th); s = __sinf(th);
    } else { c = 1.0f; s = 0.0f; }

    size_t qkRow = ((size_t)(b * HH + h) * TT + t) * DD;

    float x1 = base[i], x2 = base[i + 32];
    qb[qkRow + i]      = f2bf(c * x1 - s * x2);
    qb[qkRow + i + 32] = f2bf(s * x1 + c * x2);

    x1 = base[EE + i]; x2 = base[EE + i + 32];
    kb[qkRow + i]      = f2bf(c * x1 - s * x2);
    kb[qkRow + i + 32] = f2bf(s * x1 + c * x2);

    x1 = base[2 * EE + i]; x2 = base[2 * EE + i + 32];
    vb[((size_t)(b * HH + h) * DD + i)      * TT + t] = f2bf(x1);
    vb[((size_t)(b * HH + h) * DD + i + 32) * TT + t] = f2bf(x2);
}

// ---------------------------------------------------------------------------
// Flash-style causal attention. One wave per (b, h, 32-query tile):
// two 16-row query subtiles share every K and V fragment (2x reuse).
// Per 32-key block: 8 WMMAs for S, online softmax with half-wave shfl_xor
// row reductions, P re-swizzled C-layout -> A-layout through per-wave LDS
// tiles, then 8 WMMAs for O += P*V. Output bf16 [B,T,E].
// ---------------------------------------------------------------------------
__global__ __launch_bounds__(128) void attn_kernel(
    const unsigned short* __restrict__ qb,
    const unsigned short* __restrict__ kb,
    const unsigned short* __restrict__ vb,
    unsigned short* __restrict__ ctxb) {
    __shared__ unsigned short pbuf[4][2][16 * 32];   // [wave][qsub] P tiles

    const int lane  = threadIdx.x & 31;
    const int wave  = threadIdx.x >> 5;
    const int wid   = blockIdx.x * 4 + wave;
    const int qt = wid & 63;                // T/32 = 64 query tiles
    const int h  = (wid >> 6) & (HH - 1);
    const int b  = wid >> 10;
    const int khalf = lane >> 4;
    const int lm    = lane & 15;

    const unsigned short* qbase = qb + (size_t)(b * HH + h) * TT * DD;
    const unsigned short* kbase = kb + (size_t)(b * HH + h) * TT * DD;
    const unsigned short* vbase = vb + (size_t)(b * HH + h) * DD * TT;

    // Q A-fragments: [qsub][d-chunk], d-chunks [0,32) and [32,64)
    Frag aq[2][2];
#pragma unroll
    for (int qs = 0; qs < 2; ++qs) {
        const unsigned short* qrow =
            qbase + (size_t)(qt * 32 + qs * 16 + lm) * DD;
#pragma unroll
        for (int c = 0; c < 2; ++c)
#pragma unroll
            for (int j = 0; j < 8; ++j) {
                int d = c * 32 + ((j & 3) * 2) + ((j >> 2) * 16) + khalf * 8;
                aq[qs][c].u[j] = *(const unsigned int*)(qrow + d);
            }
    }

    float mrow[2][8], lrow[2][8];
    v8f O[2][4] = {};
#pragma unroll
    for (int qs = 0; qs < 2; ++qs)
#pragma unroll
        for (int r = 0; r < 8; ++r) { mrow[qs][r] = -1e30f; lrow[qs][r] = 0.0f; }

    const int qmax = qt * 32 + 31;
    const int nkb  = qmax / 32 + 1;

    for (int jb = 0; jb < nkb; ++jb) {
        // ---- K fragments, shared by both query subtiles ---------------
        Frag bk[2][2];                       // [ktile][d-chunk]
#pragma unroll
        for (int t2 = 0; t2 < 2; ++t2) {
            const unsigned short* krow =
                kbase + (size_t)(jb * 32 + t2 * 16 + lm) * DD;
#pragma unroll
            for (int c = 0; c < 2; ++c)
#pragma unroll
                for (int j = 0; j < 8; ++j) {
                    int d = c * 32 + ((j & 3) * 2) + ((j >> 2) * 16) + khalf * 8;
                    bk[t2][c].u[j] = *(const unsigned int*)(krow + d);
                }
        }

        // ---- scores + online softmax per query subtile ----------------
#pragma unroll
        for (int qs = 0; qs < 2; ++qs) {
            v8f s[2] = {};
#pragma unroll
            for (int t2 = 0; t2 < 2; ++t2) {
                s[t2] = __builtin_amdgcn_wmma_f32_16x16x32_bf16(
                    false, aq[qs][0].v, false, bk[t2][0].v, (short)0, s[t2],
                    false, false);
                s[t2] = __builtin_amdgcn_wmma_f32_16x16x32_bf16(
                    false, aq[qs][1].v, false, bk[t2][1].v, (short)0, s[t2],
                    false, false);
            }
#pragma unroll
            for (int r = 0; r < 8; ++r) {
                int m  = r + khalf * 8;
                int qg = qt * 32 + qs * 16 + m;
                float v0 = s[0][r] * 0.125f;           // 1/sqrt(64)
                float v1 = s[1][r] * 0.125f;
                if (jb * 32 + lm      > qg) v0 = -1e30f;
                if (jb * 32 + 16 + lm > qg) v1 = -1e30f;

                float x = fmaxf(v0, v1);
                x = fmaxf(x, __shfl_xor(x, 1));
                x = fmaxf(x, __shfl_xor(x, 2));
                x = fmaxf(x, __shfl_xor(x, 4));
                x = fmaxf(x, __shfl_xor(x, 8));

                float mnew = fmaxf(mrow[qs][r], x);
                float corr = __expf(mrow[qs][r] - mnew);
                mrow[qs][r] = mnew;

                float p0 = __expf(v0 - mnew);
                float p1 = __expf(v1 - mnew);
                float rs = p0 + p1;
                rs += __shfl_xor(rs, 1);
                rs += __shfl_xor(rs, 2);
                rs += __shfl_xor(rs, 4);
                rs += __shfl_xor(rs, 8);
                lrow[qs][r] = lrow[qs][r] * corr + rs;

#pragma unroll
                for (int t = 0; t < 4; ++t) O[qs][t][r] *= corr;

                pbuf[wave][qs][m * 32 + lm]      = f2bf(p0);
                pbuf[wave][qs][m * 32 + 16 + lm] = f2bf(p1);
            }
        }

        // LDS store->load reswizzle: DS ops are in-order per wave; wait anyway.
        asm volatile("s_wait_dscnt 0x0" ::: "memory");

        Frag ap[2];
#pragma unroll
        for (int qs = 0; qs < 2; ++qs)
#pragma unroll
            for (int j = 0; j < 8; ++j) {
                int k = ((j & 3) * 2) + ((j >> 2) * 16) + khalf * 8;
                ap[qs].u[j] = *(const unsigned int*)&pbuf[wave][qs][lm * 32 + k];
            }

        // ---- V fragments, shared by both query subtiles ---------------
        Frag bv[4];
#pragma unroll
        for (int t = 0; t < 4; ++t) {
            const unsigned short* vrow = vbase + (size_t)(t * 16 + lm) * TT;
#pragma unroll
            for (int j = 0; j < 8; ++j) {
                int key = jb * 32 + ((j & 3) * 2) + ((j >> 2) * 16) + khalf * 8;
                bv[t].u[j] = *(const unsigned int*)(vrow + key);
            }
        }

        // ---- O += P * V ----------------------------------------------
#pragma unroll
        for (int qs = 0; qs < 2; ++qs)
#pragma unroll
            for (int t = 0; t < 4; ++t)
                O[qs][t] = __builtin_amdgcn_wmma_f32_16x16x32_bf16(
                    false, ap[qs].v, false, bv[t].v, (short)0, O[qs][t],
                    false, false);
    }

    // ---- normalize + write context bf16 [B,T,E] ----------------------
#pragma unroll
    for (int qs = 0; qs < 2; ++qs)
#pragma unroll
        for (int t = 0; t < 4; ++t)
#pragma unroll
            for (int r = 0; r < 8; ++r) {
                int row = qt * 32 + qs * 16 + r + khalf * 8;
                int e   = h * DD + t * 16 + lm;
                float val = O[qs][t][r] / lrow[qs][r];
                ctxb[((size_t)(b * TT + row)) * EE + e] = f2bf(val);
            }
}

// ---------------------------------------------------------------------------
// Host launcher
// ---------------------------------------------------------------------------
extern "C" void kernel_launch(void* const* d_in, const int* in_sizes, int n_in,
                              void* d_out, int out_size, void* d_ws, size_t ws_size,
                              hipStream_t stream) {
    const float* x    = (const float*)d_in[0];   // [B,T,E]
    const float* Wqkv = (const float*)d_in[1];   // [3E,E]
    const float* Wout = (const float*)d_in[2];   // [E,E]
    float* out = (float*)d_out;                  // [B,T,E]

    char* ws = (char*)d_ws;
    size_t off = 0;
    float*          qkvF = (float*)(ws + off);          off += (size_t)MM * FF * 4;
    unsigned short* xb   = (unsigned short*)(ws + off); off += (size_t)MM * EE * 2;
    unsigned short* wqb  = (unsigned short*)(ws + off); off += (size_t)FF * EE * 2;
    unsigned short* wob  = (unsigned short*)(ws + off); off += (size_t)EE * EE * 2;
    unsigned short* qb   = (unsigned short*)(ws + off); off += (size_t)MM * EE * 2;
    unsigned short* kb   = (unsigned short*)(ws + off); off += (size_t)MM * EE * 2;
    unsigned short* vb   = (unsigned short*)(ws + off); off += (size_t)MM * EE * 2;
    unsigned short* ctxb = (unsigned short*)(ws + off); off += (size_t)MM * EE * 2;
    (void)ws_size; (void)in_sizes; (void)n_in; (void)out_size;

    // 1) fp32 -> bf16 converts
    {
        int n = MM * EE;
        cvt_bf16_kernel<<<(n + 255) / 256, 256, 0, stream>>>(x, xb, n);
        n = FF * EE;
        cvt_bf16_kernel<<<(n + 255) / 256, 256, 0, stream>>>(Wqkv, wqb, n);
        n = EE * EE;
        cvt_bf16_kernel<<<(n + 255) / 256, 256, 0, stream>>>(Wout, wob, n);
    }

    // 2) QKV projection: [4096,3072] = xb[4096,1024] * Wqkv^T
    {
        int waves = (MM / 64) * (FF / 64);          // 3072
        gemm_bf16_kernel<<<waves / 8, 256, 0, stream>>>(xb, wqb, qkvF, MM, FF, EE);
    }

    // 3) RoPE + head split (+ V transpose)
    {
        int n = BB * TT * HH * 32;                  // 2,097,152
        rope_split_kernel<<<n / 256, 256, 0, stream>>>(qkvF, qb, kb, vb);
    }

    // 4) causal flash attention -> ctxb bf16
    {
        int waves = BB * HH * (TT / 32);            // 2048
        attn_kernel<<<waves / 4, 128, 0, stream>>>(qb, kb, vb, ctxb);
    }

    // 5) output projection: out[4096,1024] = ctxb * Wout^T
    {
        int waves = (MM / 64) * (EE / 64);          // 1024
        gemm_bf16_kernel<<<waves / 8, 256, 0, stream>>>(ctxb, wob, out, MM, EE, EE);
    }
}
